// EdgeConv_21930103013847
// MI455X (gfx1250) — matrix operand order repaired and checked by reading, not compiled, not run on hardware
//
#include <hip/hip_runtime.h>
#include <hip/hip_bf16.h>
#include <stdint.h>

typedef __attribute__((ext_vector_type(16))) _Float16 v16h;
typedef __attribute__((ext_vector_type(8)))  _Float16 v8h;
typedef __attribute__((ext_vector_type(8)))  float    v8f;
typedef __attribute__((ext_vector_type(4)))  unsigned int v4u;
typedef __attribute__((ext_vector_type(8)))  int v8i;
typedef __attribute__((ext_vector_type(4)))  int v4i;

// problem sizes
#define NCLOUD 1024
#define NPTS   64
#define CCH    64
#define KNNK   16

// padded LDS row strides (in halfs): +8 halfs (16B) per row kills bank conflicts
#define W1S 136
#define W2S 136
#define WRS 72
#define XS  72
#define H1S 136

// LDS byte offsets
#define OFF_W1   0u          // 128*136*2 = 34816
#define OFF_W2   34816u      // 34816
#define OFF_WRES 69632u      // 128*72*2 = 18432
#define OFF_XF   88064u      // 64*72*2  = 9216
#define OFF_DIST 97280u      // 64*64*4  = 16384
#define OFF_KNN  113664u     // 64*16*4  = 4096
#define OFF_MB   117760u     // 64*4 -> 256
#define OFF_DEN  118016u     // 64*4 -> 256
#define OFF_NV   118272u     // 64
#define OFF_MASK 118336u     // 64
#define OFF_XRES 118400u     // 64*128*4 = 32768
#define OFF_H1   151168u     // 8 waves * 16*136*2 = 34816
#define LDS_BYTES 185984u

// ws (f16) layout: [0,16384) W1 | [16384,32768) W2 | [32768,40960) Wres | [40960,...) x (n*4096)
#define WS_W2  16384
#define WS_WR  32768
#define WS_X   40960
#define WS_TOTAL (40960 + NCLOUD*NPTS*CCH)

#if defined(__HIP_DEVICE_COMPILE__) && __has_builtin(__builtin_amdgcn_tensor_load_to_lds) && __has_builtin(__builtin_amdgcn_s_wait_tensorcnt)
#define USE_TDM 1
#else
#define USE_TDM 0
#endif

#if USE_TDM
// Build a 2D-tile Tensor DMA Descriptor (D#) and issue TENSOR_LOAD_TO_LDS.
// dims/strides in dword units (data_size=4B). padi/pada are the encoded
// pad_interval / pad_amount codes (LDS-side padding on load).
// 6-arg builtin form (clang-23 / therock-10.0): (v4u, v8i, v4i, v4i, v8i, i32 cpol)
__device__ __forceinline__ void tdm_load_2d(unsigned lds_off, const void* g,
                                            unsigned d0, unsigned d1,
                                            unsigned t0, unsigned t1,
                                            unsigned long long s0,
                                            unsigned padi, unsigned pada) {
  unsigned long long ga = (unsigned long long)(uintptr_t)g;
  v4u g0;
  g0.x = 1u;                                   // count=1, user descriptor
  g0.y = lds_off;                              // lds_addr
  g0.z = (unsigned)(ga & 0xffffffffull);       // global_addr[31:0]
  g0.w = (unsigned)((ga >> 32) & 0x1ffffffull) | (2u << 30); // addr[56:32] | type=2
  v8i g1;
  unsigned w0 = (2u << 16)                     // data_size = 4B
              | (1u << 20)                     // pad_enable
              | (padi << 22) | (pada << 25);
  g1[0] = (int)w0;
  g1[1] = (int)((d0 & 0xffffu) << 16);                       // tensor_dim0[15:0]
  g1[2] = (int)((d0 >> 16) | ((d1 & 0xffffu) << 16));        // dim0 hi | dim1 lo
  g1[3] = (int)((d1 >> 16) | (t0 << 16));                    // dim1 hi | tile_dim0
  g1[4] = (int)(t1 & 0xffffu);                               // tile_dim1 (tile_dim2=0)
  g1[5] = (int)(s0 & 0xffffffffull);                         // dim0_stride lo
  g1[6] = (int)((s0 >> 32) & 0xffffull);                     // dim0_stride hi
  g1[7] = 0;
  v4i z4 = {0, 0, 0, 0};
  v8i z8 = {0, 0, 0, 0, 0, 0, 0, 0};
  __builtin_amdgcn_tensor_load_to_lds(g0, g1, z4, z4, z8, 0);
}
#endif

// -------- pre-kernel: convert x / W1 / W2 / Wres to f16 in workspace --------
__global__ __launch_bounds__(256) void edgeconv_cvt_f16(
    const float* __restrict__ x, const float* __restrict__ W1,
    const float* __restrict__ W2, const float* __restrict__ Wres,
    _Float16* __restrict__ ws) {
  int stride = gridDim.x * blockDim.x;
  for (int i = blockIdx.x * blockDim.x + threadIdx.x; i < WS_TOTAL; i += stride) {
    float v;
    if (i < WS_W2)      v = W1[i];
    else if (i < WS_WR) v = W2[i - WS_W2];
    else if (i < WS_X)  v = Wres[i - WS_WR];
    else                v = x[i - WS_X];
    ws[i] = (_Float16)v;
  }
}

// -------- main kernel: one workgroup per cloud --------
__global__ __launch_bounds__(256) void edgeconv_main(
    const unsigned char* __restrict__ gmask,
    const _Float16* __restrict__ ws,
    float* __restrict__ out) {
  extern __shared__ char sm[];
  _Float16* w1f = (_Float16*)(sm + OFF_W1);
  _Float16* w2f = (_Float16*)(sm + OFF_W2);
  _Float16* wrf = (_Float16*)(sm + OFF_WRES);
  _Float16* xf  = (_Float16*)(sm + OFF_XF);
  float*    dst = (float*)(sm + OFF_DIST);
  int*      knn = (int*)(sm + OFF_KNN);
  int*      mbA = (int*)(sm + OFF_MB);
  float*    denA = (float*)(sm + OFF_DEN);
  int*      nvA  = (int*)(sm + OFF_NV);
  unsigned char* maskl = (unsigned char*)(sm + OFF_MASK);
  float*    xres = (float*)(sm + OFF_XRES);
  _Float16* h1b  = (_Float16*)(sm + OFF_H1);

  const int tid  = threadIdx.x;
  const int lane = tid & 31;
  const int wave = tid >> 5;
  const int n    = blockIdx.x;
  const float INF = __builtin_inff();

  // ---- phase 0: fill LDS (TDM with LDS padding), mask, n_valid ----
#if USE_TDM
  if (wave == 0) {
    // W1: 128 rows of 64 dwords, pad 16B per 64 dwords -> LDS stride 136 halfs
    tdm_load_2d((unsigned)(uintptr_t)w1f, ws,          64, 128, 64, 128, 64, 5, 3);
    tdm_load_2d((unsigned)(uintptr_t)w2f, ws + WS_W2,  64, 128, 64, 128, 64, 5, 3);
    // Wres: 128 rows of 32 dwords, pad 16B per 32 dwords -> stride 72 halfs
    tdm_load_2d((unsigned)(uintptr_t)wrf, ws + WS_WR,  32, 128, 32, 128, 32, 4, 3);
    // x tile for this cloud: 64 rows of 32 dwords -> stride 72 halfs
    tdm_load_2d((unsigned)(uintptr_t)xf,
                ws + WS_X + (size_t)n * (NPTS * CCH), 32, 64, 32, 64, 32, 4, 3);
    __builtin_amdgcn_s_wait_tensorcnt(0);
  }
#else
  for (int i = tid; i < 128 * 128; i += 256) { int r = i >> 7, c = i & 127; w1f[r * W1S + c] = ws[i]; }
  for (int i = tid; i < 128 * 128; i += 256) { int r = i >> 7, c = i & 127; w2f[r * W2S + c] = ws[WS_W2 + i]; }
  for (int i = tid; i < 128 * 64;  i += 256) { int r = i >> 6, c = i & 63;  wrf[r * WRS + c] = ws[WS_WR + i]; }
  {
    const _Float16* xsrc = ws + WS_X + (size_t)n * (NPTS * CCH);
    for (int i = tid; i < 64 * 64; i += 256) { int r = i >> 6, c = i & 63; xf[r * XS + c] = xsrc[i]; }
  }
#endif
  if (tid < 64) maskl[tid] = gmask[(size_t)n * 64 + tid];
  if (tid == 0) {
    int s = 0;
    for (int j = 0; j < 64; ++j) s += gmask[(size_t)n * 64 + j] ? 1 : 0;
    nvA[0] = 64 - s;
  }
  __syncthreads();

  // ---- phase 1: pairwise squared distances (masked / self -> inf) ----
  for (int e = tid; e < 64 * 64; e += 256) {
    int i = e >> 6, j = e & 63;
    float dv;
    if (i == j || maskl[i] || maskl[j]) {
      dv = INF;
    } else {
      const _Float16* xi = xf + i * XS;
      const _Float16* xj = xf + j * XS;
      float acc = 0.f;
#pragma unroll 8
      for (int c = 0; c < 64; ++c) {
        float d = (float)xi[c] - (float)xj[c];
        acc = fmaf(d, d, acc);
      }
      dv = acc;
    }
    dst[i * 64 + j] = dv;
  }
  __syncthreads();

  // ---- phase 2: 16 smallest per point (lowest index on ties), mask bits ----
  if (tid < 64) {
    float* drow = dst + tid * 64;
    int*   krow = knn + tid * KNNK;
    int nv = nvA[0];
    int mb = 0, nm = 0;
    for (int kk = 0; kk < KNNK; ++kk) {
      float best = INF; int bi = 0;
      for (int j = 0; j < 64; ++j) {
        float dv = drow[j];
        if (dv < best) { best = dv; bi = j; }
      }
      drow[bi] = INF;
      krow[kk] = bi;
      if (bi >= nv) { mb |= (1 << kk); ++nm; }
    }
    mbA[tid]  = mb;
    denA[tid] = (float)(KNNK - nm);
  }
  __syncthreads();

  const int l      = lane & 15;
  const int hilane = lane >> 4;          // 0: lanes 0-15, 1: lanes 16-31
  const int koff   = hilane ? 8 : 0;     // A operand K-segment offset
  const int kboff  = hilane ? 8 : 0;     // B operand K-segment offset
  const int mrow   = hilane ? 8 : 0;     // D-layout row offset

  // ---- phase 3: x_res = x @ Wres^T via WMMA (M = points), into LDS ----
  for (int job = wave; job < 32; job += 8) {
    int mt = job >> 3, nt = job & 7;
    v8f acc = {};
#pragma unroll
    for (int kc = 0; kc < 2; ++kc) {
      const _Float16* xr = xf + (mt * 16 + l) * XS + kc * 32 + koff;
      v8h alo = *(const v8h*)xr;
      v8h ahi = *(const v8h*)(xr + 16);
      v16h a;
#pragma unroll
      for (int i = 0; i < 8; ++i) { a[i] = alo[i]; a[8 + i] = ahi[i]; }
      const _Float16* wr = wrf + (nt * 16 + l) * WRS + kc * 32 + kboff;
      v8h blo = *(const v8h*)wr;
      v8h bhi = *(const v8h*)(wr + 16);
      v16h b;
#pragma unroll
      for (int i = 0; i < 8; ++i) { b[i] = blo[i]; b[8 + i] = bhi[i]; }
      acc = __builtin_amdgcn_wmma_f32_16x16x32_f16(false, a, false, b, (short)0, acc, false, false);
    }
#pragma unroll
    for (int r = 0; r < 8; ++r)
      xres[(mt * 16 + mrow + r) * 128 + nt * 16 + l] = acc[r];
  }
  __syncthreads();

  // ---- phase 4: per-point fused feat -> GEMM1 -> GEMM2 -> mean -> out ----
  _Float16* h1w = h1b + wave * (16 * H1S);
  for (int p = wave; p < 64; p += 8) {
    const _Float16* xrow = xf + p * XS;
    const int* krow = knn + p * KNNK;
    // reference quirk: f_neighbor is the scalar channel x[p][knn_j]
    _Float16 s = xrow[krow[l]];

    // GEMM1: feat(16x128) @ W1^T -> h1(16x128), f32 accum
    v8f acc[8] = {};
#pragma unroll
    for (int kc = 0; kc < 4; ++kc) {
      v16h a;
      if (kc < 2) {                        // center half: feat[c] = x[p][c]
        const _Float16* xr = xrow + kc * 32 + koff;
        v8h lo = *(const v8h*)xr;
        v8h hi = *(const v8h*)(xr + 16);
#pragma unroll
        for (int i = 0; i < 8; ++i) { a[i] = lo[i]; a[8 + i] = hi[i]; }
      } else {                             // diff half: feat[64+c] = x[p][c] - s
        const _Float16* xr = xrow + (kc - 2) * 32 + koff;
        v8h lo = *(const v8h*)xr;
        v8h hi = *(const v8h*)(xr + 16);
#pragma unroll
        for (int i = 0; i < 8; ++i) { a[i] = lo[i] - s; a[8 + i] = hi[i] - s; }
      }
#pragma unroll
      for (int nt = 0; nt < 8; ++nt) {
        const _Float16* wr = w1f + (nt * 16 + l) * W1S + kc * 32 + kboff;
        v8h blo = *(const v8h*)wr;
        v8h bhi = *(const v8h*)(wr + 16);
        v16h b;
#pragma unroll
        for (int i = 0; i < 8; ++i) { b[i] = blo[i]; b[8 + i] = bhi[i]; }
        acc[nt] = __builtin_amdgcn_wmma_f32_16x16x32_f16(false, a, false, b, (short)0, acc[nt], false, false);
      }
    }
    // relu -> f16, restage through LDS (D layout -> A layout)
#pragma unroll
    for (int nt = 0; nt < 8; ++nt)
#pragma unroll
      for (int r = 0; r < 8; ++r)
        h1w[(mrow + r) * H1S + nt * 16 + l] = (_Float16)fmaxf(acc[nt][r], 0.f);
    asm volatile("s_wait_dscnt 0" ::: "memory");

    // GEMM2: h1(16x128) @ W2^T -> h2(16x128)
    v8f acc2[8] = {};
#pragma unroll
    for (int kc = 0; kc < 4; ++kc) {
      const _Float16* hr = h1w + l * H1S + kc * 32 + koff;
      v8h lo = *(const v8h*)hr;
      v8h hi = *(const v8h*)(hr + 16);
      v16h a;
#pragma unroll
      for (int i = 0; i < 8; ++i) { a[i] = lo[i]; a[8 + i] = hi[i]; }
#pragma unroll
      for (int nt = 0; nt < 8; ++nt) {
        const _Float16* wr = w2f + (nt * 16 + l) * W2S + kc * 32 + kboff;
        v8h blo = *(const v8h*)wr;
        v8h bhi = *(const v8h*)(wr + 16);
        v16h b;
#pragma unroll
        for (int i = 0; i < 8; ++i) { b[i] = blo[i]; b[8 + i] = bhi[i]; }
        acc2[nt] = __builtin_amdgcn_wmma_f32_16x16x32_f16(false, a, false, b, (short)0, acc2[nt], false, false);
      }
    }

    // relu, masked mean over the 16 neighbor rows, + x_res, relu, store
    int mb = mbA[p];
    float rde = 1.0f / denA[p];
    const float* xrp = xres + p * 128;
    float* orow = out + ((size_t)n * 64 + p) * 128;
#pragma unroll
    for (int nt = 0; nt < 8; ++nt) {
      float partial = 0.f;
#pragma unroll
      for (int r = 0; r < 8; ++r) {
        int j = mrow + r;
        float v = fmaxf(acc2[nt][r], 0.f);
        partial += ((mb >> j) & 1) ? 0.f : v;
      }
      float tot = partial + __shfl_xor(partial, 16, 32);
      float res = fmaxf(tot * rde + xrp[nt * 16 + l], 0.f);
      if (hilane == 0) orow[nt * 16 + l] = res;
    }
  }
}

extern "C" void kernel_launch(void* const* d_in, const int* in_sizes, int n_in,
                              void* d_out, int out_size, void* d_ws, size_t ws_size,
                              hipStream_t stream) {
  (void)in_sizes; (void)n_in; (void)out_size; (void)ws_size;
  const float* x            = (const float*)d_in[0];
  const unsigned char* mask = (const unsigned char*)d_in[1];
  const float* W1           = (const float*)d_in[2];
  const float* W2           = (const float*)d_in[3];
  const float* Wres         = (const float*)d_in[4];
  _Float16* ws = (_Float16*)d_ws;
  float* out = (float*)d_out;

  edgeconv_cvt_f16<<<512, 256, 0, stream>>>(x, W1, W2, Wres, ws);
  edgeconv_main<<<NCLOUD, 256, LDS_BYTES, stream>>>(mask, ws, out);
}
